// ReferenceMLP_16217796509889
// MI455X (gfx1250) — matrix operand order repaired and checked by reading, not compiled, not run on hardware
//
#include <hip/hip_runtime.h>
#include <cstdint>
#include <cstddef>

// ---------------------------------------------------------------------------
// MoE dense-expert MLP for MI455X (gfx1250, wave32, WMMA + async-to-LDS).
//   B=2 S=1024 -> T=2048 tokens, H=768, E=16 experts, F=768, top-k=4
// Compute-bound (~116 GFLOP): both big GEMMs run on v_wmma_f32_16x16x32_f16.
// A-tiles (already f16 in memory) are staged with GLOBAL_LOAD_ASYNC_TO_LDS_B128
// (ASYNCcnt-tracked, no VGPR round-trip); B-tiles go through VGPRs because they
// need f32->f16 conversion + transpose.
// Workspace layout (needs ~54 MB):
//   [0]                       float   scores[T][E]          (128 KB)
//   [+T*E*4]                  f16     xh[T][H]              (3 MB)
//   [+T*H*2]                  f16     act[E][T][F]          (48 MB)
// ---------------------------------------------------------------------------

#define MOE_ALPHA 1.702f
#define MOE_LIMIT 7.0f

constexpr int Bb = 2, Ss = 1024, Hh = 768, Ee = 16, Ff = 768;
constexpr int Tt = Bb * Ss;           // 2048 tokens

typedef __attribute__((ext_vector_type(16))) _Float16 v16h;
typedef __attribute__((ext_vector_type(8)))  _Float16 v8h;
typedef __attribute__((ext_vector_type(8)))  float    v8f;

union FragA { v16h v; v8h h[2]; };

// Low 32 bits of a flat pointer into LDS == wave-relative LDS byte address
// (CDNA5 aperture rule).
__device__ __forceinline__ uint32_t lds_addr32(const void* p) {
    return (uint32_t)(uintptr_t)p;
}

// 32 contiguous bytes global -> LDS, async (two b128, INST_OFFSET advances
// both the LDS and global addresses). Tracked on ASYNCcnt.
__device__ __forceinline__ void async_copy32(uint32_t lds_off, const void* gptr) {
    asm volatile(
        "global_load_async_to_lds_b128 %0, %1, off\n\t"
        "global_load_async_to_lds_b128 %0, %1, off offset:16"
        :: "v"(lds_off), "v"((uint64_t)(uintptr_t)gptr) : "memory");
}

__device__ __forceinline__ void wait_async0() {
    asm volatile("s_wait_asynccnt 0" ::: "memory");
}

// ---------------------------------------------------------------- utilities
__global__ __launch_bounds__(256) void zero_out(float* __restrict__ p, int n) {
    int i = blockIdx.x * 256 + threadIdx.x;
    if (i < n) p[i] = 0.0f;
}

__global__ __launch_bounds__(256) void cast_x(const float* __restrict__ x,
                                              _Float16* __restrict__ xh, int n) {
    int i = blockIdx.x * 256 + threadIdx.x;
    if (i < n) xh[i] = (_Float16)x[i];
}

// ---------------------------------------------------------------- router
// One wave per token. Lanes 0..15 own one expert each; the two lane-halves
// split H. Top-4 + softmax done redundantly per lane (16 values only).
__global__ __launch_bounds__(128)
void router_kernel(const float* __restrict__ x,      // [T,H]
                   const float* __restrict__ rw,     // [E,H]
                   const float* __restrict__ rb,     // [E]
                   float* __restrict__ scores,       // ws  [T,E]
                   float* __restrict__ scores_out) { // out [T,E]
    const int wave = threadIdx.x >> 5;
    const int lane = threadIdx.x & 31;
    const int t    = blockIdx.x * 4 + wave;
    const int ex   = lane & 15;
    const int half = lane >> 4;

    const float* xr = x  + (size_t)t  * Hh + half * (Hh / 2);
    const float* wr = rw + (size_t)ex * Hh + half * (Hh / 2);
    float s = 0.0f;
    #pragma unroll 4
    for (int h = 0; h < Hh / 2; h += 4) {
        float4 a = *(const float4*)(xr + h);
        float4 b = *(const float4*)(wr + h);
        s += a.x * b.x + a.y * b.y + a.z * b.z + a.w * b.w;
    }
    s += __shfl_xor(s, 16, 32);   // combine H halves
    s += rb[ex];                  // logit for expert `ex` (both halves equal)

    float v[16];
    #pragma unroll
    for (int i = 0; i < 16; ++i) v[i] = __shfl(s, i, 32);

    // top-4 (strict > keeps lowest index on ties, like jax.lax.top_k)
    float sc[16]; bool used[16];
    for (int i = 0; i < 16; ++i) { sc[i] = 0.0f; used[i] = false; }
    float val[4]; int idx[4];
    for (int k = 0; k < 4; ++k) {
        float best = -3.4e38f; int bi = 0;
        for (int i = 0; i < 16; ++i)
            if (!used[i] && v[i] > best) { best = v[i]; bi = i; }
        used[bi] = true; val[k] = best; idx[k] = bi;
    }
    const float mx = val[0];
    float p[4], sum = 0.0f;
    for (int k = 0; k < 4; ++k) { p[k] = __expf(val[k] - mx); sum += p[k]; }
    const float inv = 1.0f / sum;
    for (int k = 0; k < 4; ++k) sc[idx[k]] = p[k] * inv;

    if (lane < 16) {
        scores[(size_t)t * Ee + lane]     = sc[lane];
        scores_out[(size_t)t * Ee + lane] = sc[lane];
    }
}

// ---------------------------------------------------------------- GEMM1+GLU
// Tile: 128 tokens x 64 F-cols per block, one expert per blockIdx.z.
// 8 waves as 4(M) x 2(N); each wave owns a 32x32 C tile for gate AND up
// (2x2 wmma accumulators each). K loop over H in steps of 32.
__global__ __launch_bounds__(256)
void gemm1_act(const _Float16* __restrict__ xh,    // [T,H] f16
               const float* __restrict__ gup,      // [E,H,2F] interleaved gate/up
               const float* __restrict__ gup_bias, // [E,2F]
               _Float16* __restrict__ act) {       // [E,T,F] f16
    const int e  = blockIdx.z;
    const int t0 = blockIdx.y * 128;
    const int f0 = blockIdx.x * 64;

    __shared__ alignas(32) _Float16 As[128][32]; // [row][k]
    __shared__ alignas(32) _Float16 Bg[64][32];  // [col][k]  (transposed)
    __shared__ alignas(32) _Float16 Bu[64][32];

    const int tid  = threadIdx.x;
    const int lane = tid & 31;
    const int wave = tid >> 5;
    const int wm   = (wave & 3) * 32;   // wave M offset inside block tile
    const int wn   = (wave >> 2) * 32;  // wave N offset inside block tile

    v8f accg[2][2], accu[2][2];
    #pragma unroll
    for (int i = 0; i < 2; ++i)
        #pragma unroll
        for (int j = 0; j < 2; ++j)
            #pragma unroll
            for (int q = 0; q < 8; ++q) { accg[i][j][q] = 0.0f; accu[i][j][q] = 0.0f; }

    const int ar  = tid >> 1;           // A tile row this thread fills
    const int ac  = (tid & 1) * 16;     // 16 contiguous k per thread
    const int bf  = tid & 63;           // B tile column (f)
    const int bk0 = (tid >> 6) * 8;     // 8 k rows per thread

    const int arow = lane & 15;         // wmma A: row per lane
    const int koff = (lane >> 4) * 8;   // wmma A: k base per lane-half
    const int kb   = (lane >> 4) * 8;   // wmma B: k base per lane-half

    const uint32_t a_lds = lds_addr32(&As[ar][ac]);

    for (int k0 = 0; k0 < Hh; k0 += 32) {
        __syncthreads();
        // A tile: async global->LDS (f16, no conversion needed)
        async_copy32(a_lds, xh + (size_t)(t0 + ar) * Hh + (k0 + ac));
        { // B tiles: de-interleave gate/up and transpose into [col][k]
            const float* base = gup + ((size_t)e * Hh + (k0 + bk0)) * (2 * Ff)
                                    + 2 * (size_t)(f0 + bf);
            if (k0 + 32 < Hh)   // stream next k-step's weight rows toward L2
                __builtin_prefetch(base + (size_t)32 * (2 * Ff), 0, 0);
            #pragma unroll
            for (int i = 0; i < 8; ++i) {
                float2 w = *(const float2*)(base + (size_t)i * (2 * Ff));
                Bg[bf][bk0 + i] = (_Float16)w.x;   // gate = col 2f
                Bu[bf][bk0 + i] = (_Float16)w.y;   // up   = col 2f+1
            }
        }
        wait_async0();
        __syncthreads();

        FragA a[2];
        #pragma unroll
        for (int mi = 0; mi < 2; ++mi) {
            const int r = wm + mi * 16 + arow;
            a[mi].h[0] = *(const v8h*)&As[r][koff];       // K = koff..koff+7
            a[mi].h[1] = *(const v8h*)&As[r][16 + koff];  // K = 16+koff..
        }
        #pragma unroll
        for (int ni = 0; ni < 2; ++ni) {
            const int c = wn + ni * 16 + (lane & 15);
            FragA bg, bu;
            bg.h[0] = *(const v8h*)&Bg[c][kb];
            bg.h[1] = *(const v8h*)&Bg[c][kb + 16];
            bu.h[0] = *(const v8h*)&Bu[c][kb];
            bu.h[1] = *(const v8h*)&Bu[c][kb + 16];
            #pragma unroll
            for (int mi = 0; mi < 2; ++mi) {
                accg[mi][ni] = __builtin_amdgcn_wmma_f32_16x16x32_f16(
                    false, a[mi].v, false, bg.v, (short)0, accg[mi][ni], false, false);
                accu[mi][ni] = __builtin_amdgcn_wmma_f32_16x16x32_f16(
                    false, a[mi].v, false, bu.v, (short)0, accu[mi][ni], false, false);
            }
        }
    }

    // epilogue: bias + clamp + GLU, store f16 activation
    #pragma unroll
    for (int ni = 0; ni < 2; ++ni) {
        const int fc = f0 + wn + ni * 16 + (lane & 15);
        float2 bias = *(const float2*)(gup_bias + (size_t)e * 2 * Ff + 2 * (size_t)fc);
        #pragma unroll
        for (int mi = 0; mi < 2; ++mi) {
            #pragma unroll
            for (int i = 0; i < 8; ++i) {
                const int tr = t0 + wm + mi * 16 + i + ((lane >> 4) * 8);
                float g = fminf(accg[mi][ni][i] + bias.x, MOE_LIMIT);
                float u = fminf(fmaxf(accu[mi][ni][i] + bias.y, -MOE_LIMIT), MOE_LIMIT);
                float glu = g / (1.0f + __expf(-MOE_ALPHA * g));
                act[((size_t)e * Tt + tr) * Ff + fc] = (_Float16)((u + 1.0f) * glu);
            }
        }
    }
}

// ---------------------------------------------------------------- GEMM2+mix
// y[e] = act[e] @ down[e]; out[t,h] += w[t,e] * (y + bias_e[h])  (atomic add)
__global__ __launch_bounds__(256)
void gemm2_combine(const _Float16* __restrict__ act, // [E,T,F] f16
                   const float* __restrict__ dw,     // [E,F,H]
                   const float* __restrict__ dbias,  // [E,H]
                   const float* __restrict__ scores, // [T,E]
                   float* __restrict__ out) {        // [T,H]
    const int e  = blockIdx.z;
    const int t0 = blockIdx.y * 128;
    const int h0 = blockIdx.x * 64;

    __shared__ alignas(32) _Float16 As[128][32];
    __shared__ alignas(32) _Float16 Bs[64][32];
    __shared__ float sw[128];

    const int tid  = threadIdx.x;
    const int lane = tid & 31;
    const int wave = tid >> 5;
    const int wm   = (wave & 3) * 32;
    const int wn   = (wave >> 2) * 32;

    if (tid < 128) sw[tid] = scores[(size_t)(t0 + tid) * Ee + e];

    v8f acc[2][2];
    #pragma unroll
    for (int i = 0; i < 2; ++i)
        #pragma unroll
        for (int j = 0; j < 2; ++j)
            #pragma unroll
            for (int q = 0; q < 8; ++q) acc[i][j][q] = 0.0f;

    const int ar  = tid >> 1;
    const int ac  = (tid & 1) * 16;
    const int bh  = tid & 63;
    const int bk0 = (tid >> 6) * 8;
    const int arow = lane & 15;
    const int koff = (lane >> 4) * 8;
    const int kb   = (lane >> 4) * 8;

    const uint32_t a_lds = lds_addr32(&As[ar][ac]);

    for (int k0 = 0; k0 < Ff; k0 += 32) {
        __syncthreads();
        // A: act[e][t0+ar][k0+ac..], async global->LDS
        async_copy32(a_lds, act + ((size_t)e * Tt + (t0 + ar)) * Ff + (k0 + ac));
        { // B: down[e][k0+bk0+i][h0+bh] -> Bs[h][k] (transpose + f32->f16)
            const float* base = dw + ((size_t)e * Ff + (k0 + bk0)) * Hh + (h0 + bh);
            if (k0 + 32 < Ff)
                __builtin_prefetch(base + (size_t)32 * Hh, 0, 0);
            #pragma unroll
            for (int i = 0; i < 8; ++i)
                Bs[bh][bk0 + i] = (_Float16)base[(size_t)i * Hh];
        }
        wait_async0();
        __syncthreads();

        FragA a[2];
        #pragma unroll
        for (int mi = 0; mi < 2; ++mi) {
            const int r = wm + mi * 16 + arow;
            a[mi].h[0] = *(const v8h*)&As[r][koff];
            a[mi].h[1] = *(const v8h*)&As[r][16 + koff];
        }
        #pragma unroll
        for (int ni = 0; ni < 2; ++ni) {
            const int c = wn + ni * 16 + (lane & 15);
            FragA b;
            b.h[0] = *(const v8h*)&Bs[c][kb];
            b.h[1] = *(const v8h*)&Bs[c][kb + 16];
            #pragma unroll
            for (int mi = 0; mi < 2; ++mi)
                acc[mi][ni] = __builtin_amdgcn_wmma_f32_16x16x32_f16(
                    false, a[mi].v, false, b.v, (short)0, acc[mi][ni], false, false);
        }
    }

    #pragma unroll
    for (int ni = 0; ni < 2; ++ni) {
        const int hc = h0 + wn + ni * 16 + (lane & 15);
        const float bias = dbias[(size_t)e * Hh + hc];
        #pragma unroll
        for (int mi = 0; mi < 2; ++mi) {
            #pragma unroll
            for (int i = 0; i < 8; ++i) {
                const int rloc = wm + mi * 16 + i + ((lane >> 4) * 8);
                const float w = sw[rloc];
                const int tr = t0 + rloc;
                atomicAdd(&out[(size_t)tr * Hh + hc], w * (acc[mi][ni][i] + bias));
            }
        }
    }
}

// ---------------------------------------------------------------- launcher
extern "C" void kernel_launch(void* const* d_in, const int* in_sizes, int n_in,
                              void* d_out, int out_size, void* d_ws, size_t ws_size,
                              hipStream_t stream) {
    const float* x      = (const float*)d_in[0];  // [B,S,H]
    const float* rw     = (const float*)d_in[1];  // [E,H]
    const float* rb     = (const float*)d_in[2];  // [E]
    const float* gup    = (const float*)d_in[3];  // [E,H,2F]
    const float* gup_b  = (const float*)d_in[4];  // [E,2F]
    const float* dwn    = (const float*)d_in[5];  // [E,F,H]
    const float* dwn_b  = (const float*)d_in[6];  // [E,H]
    (void)in_sizes; (void)n_in; (void)out_size; (void)ws_size;

    float* out        = (float*)d_out;            // routed_out [T,H]
    float* scores_out = out + (size_t)Tt * Hh;    // router_scores [T,E]

    char* ws = (char*)d_ws;
    float*    scores = (float*)ws;                                   // T*E f32
    _Float16* xh     = (_Float16*)(ws + (size_t)Tt * Ee * 4);        // T*H f16
    _Float16* act    = (_Float16*)(ws + (size_t)Tt * Ee * 4
                                      + (size_t)Tt * Hh * 2);        // E*T*F f16

    const int nOut = Tt * Hh;
    zero_out<<<(nOut + 255) / 256, 256, 0, stream>>>(out, nOut);
    cast_x  <<<(nOut + 255) / 256, 256, 0, stream>>>(x, xh, nOut);
    router_kernel<<<Tt / 4, 128, 0, stream>>>(x, rw, rb, scores, scores_out);

    dim3 g1(Ff / 64, Tt / 128, Ee);   // 12 x 16 x 16
    gemm1_act<<<g1, 256, 0, stream>>>(xh, gup, gup_b, act);

    dim3 g2(Hh / 64, Tt / 128, Ee);   // 12 x 16 x 16
    gemm2_combine<<<g2, 256, 0, stream>>>(act, dwn, dwn_b, scores, out);
}
